// RPN_83829171683876
// MI455X (gfx1250) — matrix operand order repaired and checked by reading, not compiled, not run on hardware
//
#include <hip/hip_runtime.h>
#include <hip/hip_bf16.h>

// ---------------------------------------------------------------------------
// RPN forward for MI455X (gfx1250, wave32, WMMA).
//
// 3x3 conv 512->512 = 71.7 GFLOP vs ~40MB traffic (1.7us @ 23.3 TB/s) ->
// compute-bound -> implicit GEMM with v_wmma_f32_16x16x32_bf16 (f32 accum).
// K-loop is double-buffered in LDS with a 2-deep global->VGPR prefetch
// pipeline: one workgroup barrier per 32-K chunk, global loads for chunk k+2
// overlap the WMMAs of chunk k. 1x1 cls+bbox convs fused into an N=64 WMMA
// GEMM. Sort/NMS working set (<8MB) is L2-resident (192MB L2).
// ---------------------------------------------------------------------------

typedef __bf16 bf16_t;
typedef __attribute__((ext_vector_type(16))) __bf16 v16bf;
typedef __attribute__((ext_vector_type(8)))  __bf16 bf16x8;
typedef __attribute__((ext_vector_type(8)))  float  v8f;

struct __align__(16) U4 { unsigned x, y, z, w; };

#define H_FEAT 100
#define W_FEAT 152
#define HP     104    // 100 + 2 pad + 2 guard rows (OOB tile rows read zeros)
#define WP     156    // 152 + 2 pad + 2 guard cols
#define CCH    512
#define M_TOT  15200  // H*W
#define M_PAD  15232  // 119 * 128
#define NANCH  136800 // M_TOT * 9
#define NSORT  262144 // 2^18
#define PRE    6000
#define POST   300
#define KW188  188    // ceil(6000/32)
#define LDSTR  40     // LDS row stride in bf16 (80B) -> bank-friendly
#define ASZ    (128 * LDSTR)
#define BSZ    (64 * LDSTR)

// ---------------- phase 0: layout transforms (f32 -> bf16) -----------------

__global__ void pad_nhwc_bf16(const float* __restrict__ feat, bf16_t* __restrict__ xpad) {
  int idx = blockIdx.x * 256 + threadIdx.x;           // HP*WP*512 = 8306688
  int c = idx & 511, rest = idx >> 9;
  int w = rest % WP, h = rest / WP;
  float v = 0.0f;
  if (h >= 1 && h <= H_FEAT && w >= 1 && w <= W_FEAT)
    v = feat[((size_t)c * H_FEAT + (h - 1)) * W_FEAT + (w - 1)];
  xpad[idx] = (bf16_t)v;
}

// wt[tap][co][ci] bf16 from conv_w OIHW (tap = kh*3+kw).
__global__ void wtrans3x3(const float* __restrict__ cw, bf16_t* __restrict__ wt) {
  int idx = blockIdx.x * 256 + threadIdx.x;           // 9*512*512
  int ci = idx & 511, co = (idx >> 9) & 511, tap = idx >> 18;
  int kh = tap / 3, kw = tap % 3;
  wt[idx] = (bf16_t)cw[(((size_t)co * CCH + ci) * 3 + kh) * 3 + kw];
}

// wt2[co2][ci] bf16: 0..17 = cls_w, 18..53 = bbox_w, 54..63 = 0 (pad to N=64).
__global__ void wtrans1x1(const float* __restrict__ clsw, const float* __restrict__ bbw,
                          bf16_t* __restrict__ wt2) {
  int idx = blockIdx.x * 256 + threadIdx.x;           // 64*512
  int ci = idx & 511, co = idx >> 9;
  float v = 0.0f;
  if (co < 18)      v = clsw[(size_t)co * CCH + ci];
  else if (co < 54) v = bbw[(size_t)(co - 18) * CCH + ci];
  wt2[idx] = (bf16_t)v;
}

// ---------------- WMMA fragment loads (CDNA5 bf16 16x16x32 layouts) --------

// A 16x32 bf16: lanes0-15 M=lane: V0-3 K=0..7, V4-7 K=16..23;
//               lanes16-31:       V0-3 K=8..15, V4-7 K=24..31.
__device__ inline v16bf load_a_frag(const bf16_t* lds, int mrow, int lane) {
  int m  = mrow + (lane & 15);
  int kh = (lane >> 4) << 3;                    // 0 or 8
  bf16x8 lo = *(const bf16x8*)(lds + m * LDSTR + kh);
  bf16x8 hi = *(const bf16x8*)(lds + m * LDSTR + 16 + kh);
  v16bf r;
#pragma unroll
  for (int i = 0; i < 8; ++i) { r[i] = lo[i]; r[8 + i] = hi[i]; }
  return r;
}

// B 32x16 bf16 (LDS stored [n][k]): lanes0-15 N=lane K=0..7,16..23; lanes16-31 K=8..15,24..31.
__device__ inline v16bf load_b_frag(const bf16_t* lds, int nrow, int lane) {
  int n  = nrow + (lane & 15);
  int kk = (lane >> 4) << 3;                    // 0 or 8
  bf16x8 lo = *(const bf16x8*)(lds + n * LDSTR + kk);
  bf16x8 hi = *(const bf16x8*)(lds + n * LDSTR + 16 + kk);
  v16bf r;
#pragma unroll
  for (int i = 0; i < 8; ++i) { r[i] = lo[i]; r[8 + i] = hi[i]; }
  return r;
}

// per-chunk global source addresses for the 3x3 conv (kidx = tap*16 + kc)
__device__ inline const bf16_t* a_src3(const bf16_t* aBase, int kidx) {
  int tap = kidx >> 4, kc = kidx & 15;
  return aBase + ((size_t)(tap / 3) * WP + (tap % 3)) * CCH + kc * 32;
}
__device__ inline const bf16_t* b_src3(const bf16_t* bBase, int kidx) {
  int tap = kidx >> 4, kc = kidx & 15;
  return bBase + (size_t)tap * CCH * CCH + kc * 32;
}

// ---------------- phase 1: 3x3 conv as implicit GEMM (WMMA bf16) -----------
// Block = 8 wave32 (4Mx2N), tile 128x64, wave tile 32x32 = 2x2 wmma frags,
// 144 K-chunks of 32. Double-buffered LDS + 2-deep register prefetch:
// one barrier per chunk; chunk k+2's global loads overlap chunk k's WMMAs.
__global__ __launch_bounds__(256) void conv3x3_wmma(
    const bf16_t* __restrict__ xpad, const bf16_t* __restrict__ wt,
    const float* __restrict__ conv_b, bf16_t* __restrict__ y) {
  __shared__ __align__(16) bf16_t lA[2 * ASZ];
  __shared__ __align__(16) bf16_t lB[2 * BSZ];
  const int t = threadIdx.x, lane = t & 31, wave = t >> 5;
  const int wm = wave & 3, wn = wave >> 2;
  const int M0 = blockIdx.y * 128, N0 = blockIdx.x * 64;
  const int NCHUNK = 144;

  // A loader: thread t -> row ar (0..127), half ah (16 bf16 = 32B)
  const int ar = t >> 1, ah = t & 1;
  const int am = M0 + ar;                         // up to 15231: guard rows cover it
  const int ahh = am / W_FEAT, aww = am % W_FEAT;
  const bf16_t* aBase = xpad + ((size_t)ahh * WP + aww) * CCH + ah * 16;
  bf16_t* aDst0 = lA + ar * LDSTR + ah * 16;
  // B loader: thread t -> row bn (0..63), seg bseg (8 bf16 = 16B)
  const int bn = t >> 2, bseg = t & 3;
  const bf16_t* bBase = wt + (size_t)(N0 + bn) * CCH + bseg * 8;
  bf16_t* bDst0 = lB + bn * LDSTR + bseg * 8;

  // prologue: chunk 0 -> regs -> LDS buf0; chunk 1 -> regs
  U4 ra0, ra1, rb;
  { const bf16_t* ap = a_src3(aBase, 0);
    ra0 = *(const U4*)ap; ra1 = *(const U4*)(ap + 8);
    rb  = *(const U4*)b_src3(bBase, 0); }
  *(U4*)aDst0 = ra0; *(U4*)(aDst0 + 8) = ra1; *(U4*)bDst0 = rb;
  { const bf16_t* ap = a_src3(aBase, 1);
    ra0 = *(const U4*)ap; ra1 = *(const U4*)(ap + 8);
    rb  = *(const U4*)b_src3(bBase, 1); }
  __syncthreads();

  v8f acc[2][2] = {};
  for (int kidx = 0; kidx < NCHUNK; ++kidx) {
    const int p = kidx & 1;
    const bf16_t* As = lA + p * ASZ;
    const bf16_t* Bs = lB + p * BSZ;
    v16bf a0 = load_a_frag(As, wm * 32,      lane);
    v16bf a1 = load_a_frag(As, wm * 32 + 16, lane);
    v16bf b0 = load_b_frag(Bs, wn * 32,      lane);
    v16bf b1 = load_b_frag(Bs, wn * 32 + 16, lane);
    if (kidx + 1 < NCHUNK) {
      // stage chunk k+1 (already in regs) into the other buffer
      bf16_t* ad = aDst0 + (p ^ 1) * ASZ;
      bf16_t* bd = bDst0 + (p ^ 1) * BSZ;
      *(U4*)ad = ra0; *(U4*)(ad + 8) = ra1; *(U4*)bd = rb;
      if (kidx + 2 < NCHUNK) {
        // issue chunk k+2 global loads; they retire under the WMMAs below
        const bf16_t* ap = a_src3(aBase, kidx + 2);
        ra0 = *(const U4*)ap; ra1 = *(const U4*)(ap + 8);
        rb  = *(const U4*)b_src3(bBase, kidx + 2);
      }
    }
    acc[0][0] = __builtin_amdgcn_wmma_f32_16x16x32_bf16(false, a0, false, b0, (short)0, acc[0][0], false, false);
    acc[0][1] = __builtin_amdgcn_wmma_f32_16x16x32_bf16(false, a0, false, b1, (short)0, acc[0][1], false, false);
    acc[1][0] = __builtin_amdgcn_wmma_f32_16x16x32_bf16(false, a1, false, b0, (short)0, acc[1][0], false, false);
    acc[1][1] = __builtin_amdgcn_wmma_f32_16x16x32_bf16(false, a1, false, b1, (short)0, acc[1][1], false, false);
    __syncthreads();   // buf p^1 written; buf p reads done -> safe to reuse next iter
  }

  // epilogue: bias + ReLU + bf16 store (C layout: VGPR r -> M=r+8*(lane>=16), N=lane&15)
  const int nC = N0 + wn * 32 + (lane & 15);
  const int mB = M0 + wm * 32 + ((lane >> 4) << 3);
#pragma unroll
  for (int i = 0; i < 2; ++i)
#pragma unroll
    for (int j = 0; j < 2; ++j) {
      int n = nC + j * 16;
      float bias = conv_b[n];
#pragma unroll
      for (int r = 0; r < 8; ++r) {
        int m = mB + i * 16 + r;
        float v = acc[i][j][r] + bias;
        v = v > 0.0f ? v : 0.0f;
        if (m >= M_TOT) v = 0.0f;                 // zero the 32 pad rows of y
        y[(size_t)m * CCH + n] = (bf16_t)v;
      }
    }
}

// ---------------- phase 2: fused 1x1 cls+bbox GEMM (N=64, K=512) -----------
__global__ __launch_bounds__(256) void conv1x1_wmma(
    const bf16_t* __restrict__ y, const bf16_t* __restrict__ wt2,
    float* __restrict__ out2) {
  __shared__ __align__(16) bf16_t lA[2 * ASZ];
  __shared__ __align__(16) bf16_t lB[2 * BSZ];
  const int t = threadIdx.x, lane = t & 31, wave = t >> 5;
  const int wm = wave & 3, wn = wave >> 2;
  const int M0 = blockIdx.y * 128;
  const int NCHUNK = 16;

  const int ar = t >> 1, ah = t & 1;
  const bf16_t* aBase = y + (size_t)(M0 + ar) * CCH + ah * 16;  // y padded to 15232 rows
  bf16_t* aDst0 = lA + ar * LDSTR + ah * 16;
  const int bn = t >> 2, bseg = t & 3;
  const bf16_t* bBase = wt2 + (size_t)bn * CCH + bseg * 8;
  bf16_t* bDst0 = lB + bn * LDSTR + bseg * 8;

  U4 ra0, ra1, rb;
  ra0 = *(const U4*)aBase; ra1 = *(const U4*)(aBase + 8);
  rb  = *(const U4*)bBase;
  *(U4*)aDst0 = ra0; *(U4*)(aDst0 + 8) = ra1; *(U4*)bDst0 = rb;
  ra0 = *(const U4*)(aBase + 32); ra1 = *(const U4*)(aBase + 40);
  rb  = *(const U4*)(bBase + 32);
  __syncthreads();

  v8f acc[2][2] = {};
  for (int kidx = 0; kidx < NCHUNK; ++kidx) {
    const int p = kidx & 1;
    const bf16_t* As = lA + p * ASZ;
    const bf16_t* Bs = lB + p * BSZ;
    v16bf a0 = load_a_frag(As, wm * 32,      lane);
    v16bf a1 = load_a_frag(As, wm * 32 + 16, lane);
    v16bf b0 = load_b_frag(Bs, wn * 32,      lane);
    v16bf b1 = load_b_frag(Bs, wn * 32 + 16, lane);
    if (kidx + 1 < NCHUNK) {
      bf16_t* ad = aDst0 + (p ^ 1) * ASZ;
      bf16_t* bd = bDst0 + (p ^ 1) * BSZ;
      *(U4*)ad = ra0; *(U4*)(ad + 8) = ra1; *(U4*)bd = rb;
      if (kidx + 2 < NCHUNK) {
        int ci0 = (kidx + 2) * 32;
        ra0 = *(const U4*)(aBase + ci0); ra1 = *(const U4*)(aBase + ci0 + 8);
        rb  = *(const U4*)(bBase + ci0);
      }
    }
    acc[0][0] = __builtin_amdgcn_wmma_f32_16x16x32_bf16(false, a0, false, b0, (short)0, acc[0][0], false, false);
    acc[0][1] = __builtin_amdgcn_wmma_f32_16x16x32_bf16(false, a0, false, b1, (short)0, acc[0][1], false, false);
    acc[1][0] = __builtin_amdgcn_wmma_f32_16x16x32_bf16(false, a1, false, b0, (short)0, acc[1][0], false, false);
    acc[1][1] = __builtin_amdgcn_wmma_f32_16x16x32_bf16(false, a1, false, b1, (short)0, acc[1][1], false, false);
    __syncthreads();
  }
  const int nC = wn * 32 + (lane & 15);
  const int mB = M0 + wm * 32 + ((lane >> 4) << 3);
#pragma unroll
  for (int i = 0; i < 2; ++i)
#pragma unroll
    for (int j = 0; j < 2; ++j)
#pragma unroll
      for (int r = 0; r < 8; ++r) {
        int m = mB + i * 16 + r;
        if (m < M_TOT) out2[(size_t)m * 64 + nC + j * 16] = acc[i][j][r];
      }
}

// ---------------- phase 3: score + anchor decode + sort keys ---------------
__global__ void decode_anchors(const float* __restrict__ out2, const float* __restrict__ im_info,
                               const float* __restrict__ cls_b, const float* __restrict__ bbox_b,
                               float* __restrict__ boxes, unsigned long long* __restrict__ keys) {
  int i = blockIdx.x * 256 + threadIdx.x;               // NSORT threads
  if (i >= NSORT) return;
  if (i >= NANCH) { keys[i] = 0xFFFFFFFFFFFFFFFFull; return; }   // sort-last pad
  int m = i / 9, a = i % 9;
  int h = m / W_FEAT, w = m % W_FEAT;
  const float* row = out2 + (size_t)m * 64;
  // softmax over (bg=ch a, fg=ch 9+a) -> sigmoid of diff
  float s0 = row[a] + cls_b[a];
  float s1 = row[9 + a] + cls_b[9 + a];
  float score = 1.0f / (1.0f + expf(s0 - s1));
  float dx = row[18 + 4 * a + 0] + bbox_b[4 * a + 0];
  float dy = row[18 + 4 * a + 1] + bbox_b[4 * a + 1];
  float dw = row[18 + 4 * a + 2] + bbox_b[4 * a + 2];
  float dh = row[18 + 4 * a + 3] + bbox_b[4 * a + 3];
  // base anchor (ratios 0.5/1/2 x scales 8/16/32, base 16; rintf = np.round RNE)
  int ri = a / 3, si = a % 3;
  float ratio = ri == 0 ? 0.5f : (ri == 1 ? 1.0f : 2.0f);
  float scale = si == 0 ? 8.0f : (si == 1 ? 16.0f : 32.0f);
  float wsr = rintf(sqrtf(256.0f / ratio));
  float hsr = rintf(wsr * ratio);
  float ws2 = wsr * scale, hs2 = hsr * scale;
  float sx = (float)w * 16.0f, sy = (float)h * 16.0f;
  float ax1 = 7.5f - 0.5f * (ws2 - 1.0f) + sx;
  float ay1 = 7.5f - 0.5f * (hs2 - 1.0f) + sy;
  float ax2 = 7.5f + 0.5f * (ws2 - 1.0f) + sx;
  float ay2 = 7.5f + 0.5f * (hs2 - 1.0f) + sy;
  float aw = ax2 - ax1 + 1.0f, ahh = ay2 - ay1 + 1.0f;
  float axc = ax1 + 0.5f * aw, ayc = ay1 + 0.5f * ahh;
  float px = dx * aw + axc, py = dy * ahh + ayc;
  float pw = expf(dw) * aw, ph = expf(dh) * ahh;
  float imH = im_info[0], imW = im_info[1], sc = im_info[2];
  float x1 = fminf(fmaxf(px - 0.5f * pw, 0.0f), imW - 1.0f);
  float y1 = fminf(fmaxf(py - 0.5f * ph, 0.0f), imH - 1.0f);
  float x2 = fminf(fmaxf(px + 0.5f * pw, 0.0f), imW - 1.0f);
  float y2 = fminf(fmaxf(py + 0.5f * ph, 0.0f), imH - 1.0f);
  bool valid = ((x2 - x1 + 1.0f) >= 16.0f * sc) && ((y2 - y1 + 1.0f) >= 16.0f * sc);
  float sf = valid ? score : -1e9f;
  boxes[(size_t)i * 4 + 0] = x1; boxes[(size_t)i * 4 + 1] = y1;
  boxes[(size_t)i * 4 + 2] = x2; boxes[(size_t)i * 4 + 3] = y2;
  unsigned u  = __float_as_uint(sf);
  unsigned mv = (u & 0x80000000u) ? ~u : (u | 0x80000000u);   // ascending monotone
  keys[i] = ((unsigned long long)(~mv) << 32) | (unsigned)i;  // asc sort = desc score, asc idx
}

// ---------------- phase 4: bitonic sort (2MB keys, L2-resident) ------------
__global__ void bitonic_pass(unsigned long long* __restrict__ keys, int k, int j) {
  int i = blockIdx.x * 256 + threadIdx.x;
  int ixj = i ^ j;
  if (ixj > i) {
    unsigned long long a = keys[i], b = keys[ixj];
    bool up = ((i & k) == 0);
    if ((a > b) == up) { keys[i] = b; keys[ixj] = a; }
  }
}

__global__ void gather_top(const unsigned long long* __restrict__ keys,
                           const float* __restrict__ boxes, float* __restrict__ topb) {
  int i = blockIdx.x * 256 + threadIdx.x;
  if (i >= PRE) return;
  unsigned idx = (unsigned)(keys[i] & 0xFFFFFFFFu);
#pragma unroll
  for (int c = 0; c < 4; ++c) topb[(size_t)i * 4 + c] = boxes[(size_t)idx * 4 + c];
}

// ---------------- phase 5: NMS suppression bitmask (parallel) --------------
__global__ void nms_mask(const float* __restrict__ topb, unsigned* __restrict__ mask) {
  int gid = blockIdx.x * 256 + threadIdx.x;             // PRE * KW188
  if (gid >= PRE * KW188) return;
  int i = gid / KW188, wj = gid % KW188;
  float ix1 = topb[i * 4], iy1 = topb[i * 4 + 1], ix2 = topb[i * 4 + 2], iy2 = topb[i * 4 + 3];
  float iarea = (ix2 - ix1 + 1.0f) * (iy2 - iy1 + 1.0f);
  unsigned bits = 0;
  int j0 = wj * 32;
  for (int b = 0; b < 32; ++b) {
    int j = j0 + b;
    if (j < PRE && j > i) {
      float jx1 = topb[j * 4], jy1 = topb[j * 4 + 1], jx2 = topb[j * 4 + 2], jy2 = topb[j * 4 + 3];
      float xx1 = fmaxf(ix1, jx1), yy1 = fmaxf(iy1, jy1);
      float xx2 = fminf(ix2, jx2), yy2 = fminf(iy2, jy2);
      float inter = fmaxf(0.0f, xx2 - xx1 + 1.0f) * fmaxf(0.0f, yy2 - yy1 + 1.0f);
      float jarea = (jx2 - jx1 + 1.0f) * (jy2 - jy1 + 1.0f);
      if (inter / (iarea + jarea - inter) > 0.7f) bits |= (1u << b);
    }
  }
  mask[(size_t)i * KW188 + wj] = bits;
}

// ---------------- phase 6: greedy NMS, one wave32, wave-synchronous --------
__global__ void nms_reduce(const unsigned* __restrict__ mask, unsigned* __restrict__ keep) {
  int lane = threadIdx.x;                               // block = 32
  unsigned kw0 = ~0u, kw1 = ~0u, kw2 = ~0u, kw3 = ~0u, kw4 = ~0u, kw5 = ~0u;
  for (int i = 0; i < PRE; ++i) {
    int wi = i >> 5, owner = wi & 31, q = wi >> 5;      // q uniform
    unsigned cand;
    if (q == 0) cand = kw0; else if (q == 1) cand = kw1; else if (q == 2) cand = kw2;
    else if (q == 3) cand = kw3; else if (q == 4) cand = kw4; else cand = kw5;
    unsigned word = (unsigned)__shfl((int)cand, owner, 32);
    if ((word >> (i & 31)) & 1u) {                      // uniform branch
      const unsigned* mrow = mask + (size_t)i * KW188;
      kw0 &= ~mrow[lane];
      kw1 &= ~mrow[lane + 32];
      kw2 &= ~mrow[lane + 64];
      kw3 &= ~mrow[lane + 96];
      kw4 &= ~mrow[lane + 128];
      if (lane + 160 < KW188) kw5 &= ~mrow[lane + 160];
    }
  }
  keep[lane]       = kw0; keep[lane + 32]  = kw1; keep[lane + 64]  = kw2;
  keep[lane + 96]  = kw3; keep[lane + 128] = kw4;
  if (lane + 160 < KW188) keep[lane + 160] = kw5;
}

// ---------------- phase 7: select first 300 kept (stable) -> rois ----------
__global__ void select_rois(const unsigned* __restrict__ keep,
                            const float* __restrict__ topb, float* __restrict__ out) {
  if (threadIdx.x != 0 || blockIdx.x != 0) return;
  int cnt = 0;
  for (int i = 0; i < PRE && cnt < POST; ++i)
    if ((keep[i >> 5] >> (i & 31)) & 1u) {
      out[cnt * 5] = 0.0f;
      for (int c = 0; c < 4; ++c) out[cnt * 5 + 1 + c] = topb[i * 4 + c];
      ++cnt;
    }
  // ties at -1e9: top_k falls back to ascending index among non-kept
  for (int i = 0; i < PRE && cnt < POST; ++i)
    if (!((keep[i >> 5] >> (i & 31)) & 1u)) {
      out[cnt * 5] = 0.0f;
      for (int c = 0; c < 4; ++c) out[cnt * 5 + 1 + c] = topb[i * 4 + c];
      ++cnt;
    }
}

// ---------------------------------------------------------------------------
extern "C" void kernel_launch(void* const* d_in, const int* in_sizes, int n_in,
                              void* d_out, int out_size, void* d_ws, size_t ws_size,
                              hipStream_t stream) {
  const float* feat    = (const float*)d_in[0];
  const float* im_info = (const float*)d_in[1];
  const float* conv_w  = (const float*)d_in[2];
  const float* conv_b  = (const float*)d_in[3];
  const float* cls_w   = (const float*)d_in[4];
  const float* cls_b   = (const float*)d_in[5];
  const float* bbox_w  = (const float*)d_in[6];
  const float* bbox_b  = (const float*)d_in[7];
  float* rois = (float*)d_out;

  char* ws = (char*)d_ws;
  size_t off = 0;
  auto alloc = [&](size_t bytes) -> void* {
    void* p = (void*)(ws + off);
    off = (off + bytes + 255) & ~(size_t)255;
    return p;
  };
  bf16_t*             xpad  = (bf16_t*)alloc((size_t)HP * WP * CCH * 2);       // 16.6 MB
  bf16_t*             wt    = (bf16_t*)alloc((size_t)9 * CCH * CCH * 2);       //  4.7 MB
  bf16_t*             wt2   = (bf16_t*)alloc((size_t)64 * CCH * 2);            //  64 KB
  bf16_t*             ybuf  = (bf16_t*)alloc((size_t)M_PAD * CCH * 2);         // 15.6 MB
  float*              out2  = (float*) alloc((size_t)M_TOT * 64 * 4);          //  3.9 MB
  float*              boxes = (float*) alloc((size_t)NANCH * 4 * 4);           //  2.2 MB
  unsigned long long* keys  = (unsigned long long*)alloc((size_t)NSORT * 8);   //  2.0 MB
  float*              topb  = (float*) alloc((size_t)PRE * 4 * 4);             //  96 KB
  unsigned*           mask  = (unsigned*)alloc((size_t)PRE * KW188 * 4);       //  4.5 MB
  unsigned*           keep  = (unsigned*)alloc(256 * 4);
  (void)in_sizes; (void)n_in; (void)out_size; (void)ws_size;

  // phase 0: layout transforms
  pad_nhwc_bf16<<<(HP * WP * CCH) / 256, 256, 0, stream>>>(feat, xpad);
  wtrans3x3<<<(9 * CCH * CCH) / 256, 256, 0, stream>>>(conv_w, wt);
  wtrans1x1<<<(64 * CCH) / 256, 256, 0, stream>>>(cls_w, bbox_w, wt2);

  // phase 1: 3x3 conv implicit GEMM (compute-bound -> WMMA bf16, pipelined)
  conv3x3_wmma<<<dim3(8, 119), 256, 0, stream>>>(xpad, wt, conv_b, ybuf);

  // phase 2: fused 1x1 cls+bbox GEMM
  conv1x1_wmma<<<dim3(1, 119), 256, 0, stream>>>(ybuf, wt2, out2);

  // phase 3: score + box decode + sort keys
  decode_anchors<<<NSORT / 256, 256, 0, stream>>>(out2, im_info, cls_b, bbox_b, boxes, keys);

  // phase 4: full bitonic sort, 2^18 keys, 171 passes (array is L2-resident)
  for (int k = 2; k <= NSORT; k <<= 1)
    for (int j = k >> 1; j > 0; j >>= 1)
      bitonic_pass<<<NSORT / 256, 256, 0, stream>>>(keys, k, j);

  gather_top<<<(PRE + 255) / 256, 256, 0, stream>>>(keys, boxes, topb);

  // phases 5-7: NMS bitmask + single-wave greedy reduce + stable select
  nms_mask<<<(PRE * KW188 + 255) / 256, 256, 0, stream>>>(topb, mask);
  nms_reduce<<<1, 32, 0, stream>>>(mask, keep);
  select_rois<<<1, 1, 0, stream>>>(keep, topb, rois);
}
